// BiLSTM_CRF_74835510166005
// MI455X (gfx1250) — matrix operand order repaired and checked by reading, not compiled, not run on hardware
//
#include <hip/hip_runtime.h>
#include <hip/hip_bf16.h>

#define LSEQ 2048
#define EMB 300
#define INDIM 319
#define INSTRIDE 320
#define HID 512
#define HALFH 256
#define GATES 1024
#define QKVD 1536
#define NTAG 11
#define FSTRIDE 64
#define FFDIM 300
#define FFPAD 320
#define NEGV -10000.0f

typedef __attribute__((ext_vector_type(16))) __bf16          v16bf;
typedef __attribute__((ext_vector_type(16))) unsigned short  v16u;
typedef __attribute__((ext_vector_type(8)))  unsigned short  v8u;
typedef __attribute__((ext_vector_type(8)))  float           v8f;
typedef unsigned short ushort_t;

__device__ __forceinline__ unsigned short f2bf(float f) {
  unsigned u = __builtin_bit_cast(unsigned, f);
  u += 0x7FFFu + ((u >> 16) & 1u);   // round-to-nearest-even
  return (unsigned short)(u >> 16);
}

__device__ __forceinline__ v8f wmma_bf16(v16bf a, v16bf b, v8f c) {
  return __builtin_amdgcn_wmma_f32_16x16x32_bf16(false, a, false, b, (short)0, c, false, false);
}

// Combine two 16-byte bf16 runs into one 32-byte fragment (register placement only).
__device__ __forceinline__ v16bf load_frag(const ushort_t* p0, const ushort_t* p1) {
  v8u lo = *(const v8u*)p0;
  v8u hi = *(const v8u*)p1;
  v16u u = __builtin_shufflevector(lo, hi, 0,1,2,3,4,5,6,7,8,9,10,11,12,13,14,15);
  return __builtin_bit_cast(v16bf, u);
}

// ---------------------------------------------------------------------------
// WMMA GEMM on pre-converted bf16 operands:
//   C[M,N](f32) = scale * (A[M,K] @ B^T) (+bias[n]) (ReLU?)   B stored [N,K].
// Optional bf16 mirror output Cb. Preconditions: K%32==0, lda%8==0, ldb%8==0,
// M%32==0, N%64==0 (padded operands). One wave32 per 32x64 tile: per 32-K step
// all 12 b128 loads are issued as one clause, then 8 WMMAs consume them.
// causal!=0: skip tiles entirely above the diagonal (softmax zero-fills them).
// ---------------------------------------------------------------------------
template<bool RELU>
__global__ void gemm_wmma(const ushort_t* __restrict__ A, int lda,
                          const ushort_t* __restrict__ B, int ldb,
                          float* __restrict__ C, int ldc,
                          ushort_t* __restrict__ Cb,
                          const float* __restrict__ bias,
                          int K, float scale, int causal)
{
  const int lane = threadIdx.x & 31;
  const int hlf  = lane >> 4;
  const int l16  = lane & 15;
  const int tm = blockIdx.y * 32;
  const int tn = blockIdx.x * 64;
  if (causal && tn > tm + 31) return;   // fully-masked tile (uniform exit)

  // Persistent row pointers, advanced by +32 elems (64 B) per K-step; the
  // second 16-byte run of each fragment is a constant instruction offset.
  const ushort_t* pA0 = A + (size_t)(tm + l16) * lda      + hlf * 16;
  const ushort_t* pA1 = A + (size_t)(tm + 16 + l16) * lda + hlf * 16;
  const ushort_t* pB0 = B + (size_t)(tn      + l16) * ldb + hlf * 16;
  const ushort_t* pB1 = B + (size_t)(tn + 16 + l16) * ldb + hlf * 16;
  const ushort_t* pB2 = B + (size_t)(tn + 32 + l16) * ldb + hlf * 16;
  const ushort_t* pB3 = B + (size_t)(tn + 48 + l16) * ldb + hlf * 16;

  v8f acc[8] = {};
#pragma unroll 2
  for (int k0 = 0; k0 < K; k0 += 32) {
    // ---- issue all fragment loads for this K-step ----
    v16bf a0 = load_frag(pA0, pA0 + 8);    // A layout: 16 contiguous elems
    v16bf a1 = load_frag(pA1, pA1 + 8);
    v16bf b0 = load_frag(pB0, pB0 + 8);    // B layout: 16 contiguous elems
    v16bf b1 = load_frag(pB1, pB1 + 8);
    v16bf b2 = load_frag(pB2, pB2 + 8);
    v16bf b3 = load_frag(pB3, pB3 + 8);
    pA0 += 32; pA1 += 32; pB0 += 32; pB1 += 32; pB2 += 32; pB3 += 32;
    // ---- consume with 8 WMMAs ----
    acc[0] = wmma_bf16(a0, b0, acc[0]);
    acc[4] = wmma_bf16(a1, b0, acc[4]);
    acc[1] = wmma_bf16(a0, b1, acc[1]);
    acc[5] = wmma_bf16(a1, b1, acc[5]);
    acc[2] = wmma_bf16(a0, b2, acc[2]);
    acc[6] = wmma_bf16(a1, b2, acc[6]);
    acc[3] = wmma_bf16(a0, b3, acc[3]);
    acc[7] = wmma_bf16(a1, b3, acc[7]);
  }

#pragma unroll
  for (int nt = 0; nt < 4; ++nt) {
    int n = tn + nt * 16 + l16;
    float bv = bias ? bias[n] : 0.0f;
#pragma unroll
    for (int r = 0; r < 8; ++r) {     // C/D layout: half0 -> M=r, half1 -> M=8+r
      float v0 = acc[nt][r] * scale + bv;
      float v1 = acc[4 + nt][r] * scale + bv;
      if (RELU) { v0 = fmaxf(v0, 0.0f); v1 = fmaxf(v1, 0.0f); }
      size_t o0 = (size_t)(tm + hlf * 8 + r) * ldc + n;
      size_t o1 = (size_t)(tm + 16 + hlf * 8 + r) * ldc + n;
      C[o0] = v0; C[o1] = v1;
      if (Cb) { Cb[o0] = f2bf(v0); Cb[o1] = f2bf(v1); }
    }
  }
}

// f32 -> bf16 flat conversion
__global__ void cvt_bf16(const float* __restrict__ src, ushort_t* __restrict__ dst, int n)
{
  int i = blockIdx.x * 256 + threadIdx.x;
  if (i < n) dst[i] = f2bf(src[i]);
}

// Zero-padding copy, f32 -> bf16: dst[Rp,Cp] = bf16(src[R,C]) padded with zeros.
__global__ void pad_bf16(const float* __restrict__ src, int R, int C,
                         ushort_t* __restrict__ dst, int Rp, int Cp)
{
  int idx = blockIdx.x * 256 + threadIdx.x;
  if (idx >= Rp * Cp) return;
  int r = idx / Cp, c = idx % Cp;
  dst[idx] = (r < R && c < C) ? f2bf(src[(size_t)r * C + c]) : (ushort_t)0;
}

// Zero-padding copy, f32 -> f32 (for padded bias vectors)
__global__ void pad_f32(const float* __restrict__ src, int R, int C,
                        float* __restrict__ dst, int Rp, int Cp)
{
  int idx = blockIdx.x * 256 + threadIdx.x;
  if (idx >= Rp * Cp) return;
  int r = idx / Cp, c = idx % Cp;
  dst[idx] = (r < R && c < C) ? src[(size_t)r * C + c] : 0.0f;
}

// LDS tile transpose, f32 in -> bf16 out: dst[Cc,R] = bf16(src[R,Cc]^T)
__global__ void transpose_bf16(const float* __restrict__ src, int lds_, int R, int Cc,
                               ushort_t* __restrict__ dst, int ldd)
{
  __shared__ float tile[16][17];
  int r0 = blockIdx.y * 16, c0 = blockIdx.x * 16;
  int tx = threadIdx.x, ty = threadIdx.y;
  int r = r0 + ty, c = c0 + tx;
  tile[ty][tx] = (r < R && c < Cc) ? src[(size_t)r * lds_ + c] : 0.0f;
  __syncthreads();
  int orow = c0 + ty, ocol = r0 + tx;
  if (orow < Cc && ocol < R) dst[(size_t)orow * ldd + ocol] = f2bf(tile[tx][ty]);
}

// ---------------------------------------------------------------------------
// Build input features (bf16) X[L,320]: [embed(300) | one-hot pos(18) | tfidf | 0]
// ---------------------------------------------------------------------------
__global__ void build_input(const int* __restrict__ sentence,
                            const int* __restrict__ postag,
                            const float* __restrict__ tfidf,
                            const float* __restrict__ wembed,
                            ushort_t* __restrict__ Xb)
{
  int t = blockIdx.x, d = threadIdx.x;   // block = 320 threads
  int w = sentence[t];
  float v;
  if (d < EMB)             v = wembed[(size_t)w * EMB + d];
  else if (d < EMB + 18)   v = (postag[t] == (d - EMB)) ? 1.0f : 0.0f;
  else if (d == INDIM - 1) v = tfidf[t];
  else                     v = 0.0f;
  Xb[(size_t)t * INSTRIDE + d] = f2bf(v);
}

// ---------------------------------------------------------------------------
// LSTM direction scan (gridDim.x = 2: 0=fwd, 1=bwd). gates_in = x@Wi^T precomputed.
// 1024 threads: thread owns one gate row (dot-256 vs h); gate preacts staged in
// LDS; threads 0..255 apply the cell update. h lives in LDS across steps.
// Emits f32 hout and a bf16 mirror for the next layer's input GEMM.
// ---------------------------------------------------------------------------
__global__ void lstm_scan(const float* __restrict__ gates_f, const float* __restrict__ gates_b,
                          const float* __restrict__ Whf, const float* __restrict__ bif, const float* __restrict__ bhf,
                          const float* __restrict__ Whb, const float* __restrict__ bib, const float* __restrict__ bhb,
                          const float* __restrict__ h0, const float* __restrict__ c0,   // [2,256] layer slice
                          float* __restrict__ hout, ushort_t* __restrict__ houtb)       // [L,512]
{
  const int dir = blockIdx.x;
  const float* gin = dir ? gates_b : gates_f;
  const float* Wh  = dir ? Whb : Whf;
  const float* bi  = dir ? bib : bif;
  const float* bh  = dir ? bhb : bhf;

  __shared__ float hsh[HALFH];
  __shared__ float gsh[GATES];
  const int tid = threadIdx.x;   // 0..1023
  float c = 0.0f;
  if (tid < HALFH) {
    c = c0[dir * HALFH + tid];
    hsh[tid] = h0[dir * HALFH + tid];
  }
  __syncthreads();

  const float bsum = bi[tid] + bh[tid];
  const float* wrow = Wh + (size_t)tid * HALFH;

  for (int s = 0; s < LSEQ; ++s) {
    int t = dir ? (LSEQ - 1 - s) : s;
    float g = gin[(size_t)t * GATES + tid] + bsum;
#pragma unroll 8
    for (int k = 0; k < HALFH; ++k) g = fmaf(wrow[k], hsh[k], g);
    gsh[tid] = g;
    __syncthreads();              // all matvec reads of hsh done; gsh complete
    if (tid < HALFH) {
      float i_ = 1.0f / (1.0f + __expf(-gsh[tid]));
      float f_ = 1.0f / (1.0f + __expf(-gsh[256 + tid]));
      float gg = tanhf(gsh[512 + tid]);
      float o_ = 1.0f / (1.0f + __expf(-gsh[768 + tid]));
      c = f_ * c + i_ * gg;
      float h = o_ * tanhf(c);
      hsh[tid] = h;
      size_t o = (size_t)t * HID + dir * HALFH + tid;
      hout[o] = h;
      houtb[o] = f2bf(h);
    }
    __syncthreads();              // new h visible to all
  }
}

// src = lstm_out * sqrt(H) + positional_encoding  (f32 + bf16 mirror)
__global__ void posenc_scale(const float* __restrict__ lo,
                             float* __restrict__ src, ushort_t* __restrict__ srcb)
{
  int idx = blockIdx.x * blockDim.x + threadIdx.x;
  if (idx >= LSEQ * HID) return;
  int t = idx / HID, d = idx % HID;
  int i2 = d & ~1;
  float div = __expf((float)i2 * (-9.210340371976184f / 512.0f));
  float ang = (float)t * div;
  float pe = (d & 1) ? __cosf(ang) : __sinf(ang);
  float v = lo[idx] * 22.62741699796952f /*sqrt(512)*/ + pe;
  src[idx] = v;
  srcb[idx] = f2bf(v);
}

// Row-wise causal softmax on S[L,L] in place; emits bf16 mirror Sb. Block per row.
__global__ void softmax_causal(float* __restrict__ S, ushort_t* __restrict__ Sb)
{
  int i = blockIdx.x, tid = threadIdx.x;
  float* row = S + (size_t)i * LSEQ;
  ushort_t* rowb = Sb + (size_t)i * LSEQ;
  __shared__ float red[256];
  float mx = -1e30f;
  for (int j = tid; j <= i; j += 256) mx = fmaxf(mx, row[j]);
  red[tid] = mx; __syncthreads();
  for (int s = 128; s > 0; s >>= 1) { if (tid < s) red[tid] = fmaxf(red[tid], red[tid + s]); __syncthreads(); }
  mx = red[0]; __syncthreads();
  float sum = 0.0f;
  for (int j = tid; j < LSEQ; j += 256) {
    float v = (j <= i) ? __expf(row[j] - mx) : 0.0f;
    row[j] = v; sum += v;
  }
  red[tid] = sum; __syncthreads();
  for (int s = 128; s > 0; s >>= 1) { if (tid < s) red[tid] += red[tid + s]; __syncthreads(); }
  float inv = 1.0f / red[0];
  for (int j = tid; j < LSEQ; j += 256) {
    float v = row[j] * inv;
    row[j] = v;
    rowb[j] = f2bf(v);
  }
}

// x = LayerNorm(x + y) * g + b  (in place, f32 + bf16 mirror)
__global__ void add_layernorm(float* __restrict__ x, ushort_t* __restrict__ xb,
                              const float* __restrict__ y,
                              const float* __restrict__ g, const float* __restrict__ b)
{
  int i = blockIdx.x, tid = threadIdx.x;
  __shared__ float red[256];
  size_t base = (size_t)i * HID;
  float v0 = x[base + tid]       + y[base + tid];
  float v1 = x[base + 256 + tid] + y[base + 256 + tid];
  red[tid] = v0 + v1; __syncthreads();
  for (int s = 128; s > 0; s >>= 1) { if (tid < s) red[tid] += red[tid + s]; __syncthreads(); }
  float mean = red[0] * (1.0f / 512.0f); __syncthreads();
  float d0 = v0 - mean, d1 = v1 - mean;
  red[tid] = d0 * d0 + d1 * d1; __syncthreads();
  for (int s = 128; s > 0; s >>= 1) { if (tid < s) red[tid] += red[tid + s]; __syncthreads(); }
  float inv = rsqrtf(red[0] * (1.0f / 512.0f) + 1e-5f);
  float o0 = d0 * inv * g[tid]       + b[tid];
  float o1 = d1 * inv * g[256 + tid] + b[256 + tid];
  x[base + tid]        = o0;  xb[base + tid]        = f2bf(o0);
  x[base + 256 + tid]  = o1;  xb[base + 256 + tid]  = f2bf(o1);
}

// CRF: forward partition scan + gold score; out[0] = fwd - gold
__global__ void crf_kernel(const float* __restrict__ feats /*[L,64]*/,
                           const float* __restrict__ trans /*[11,11]*/,
                           const int* __restrict__ tags,
                           float* __restrict__ out)
{
  __shared__ float fv[16], nfv[16], red[32];
  int tid = threadIdx.x;    // block = 32 (one wave)
  float acc = 0.0f;
  for (int t = tid; t < LSEQ; t += 32) {
    int cur = tags[t];
    int prev = (t == 0) ? 0 /*START*/ : tags[t - 1];
    acc += feats[(size_t)t * FSTRIDE + cur] + trans[cur * NTAG + prev];
  }
  red[tid] = acc; __syncthreads();
  for (int s = 16; s > 0; s >>= 1) { if (tid < s) red[tid] += red[tid + s]; __syncthreads(); }
  float gold = red[0] + trans[1 * NTAG + tags[LSEQ - 1]];

  if (tid < NTAG) fv[tid] = (tid == 0) ? 0.0f : NEGV;
  __syncthreads();
  for (int t = 0; t < LSEQ; ++t) {
    if (tid < NTAG) {
      float ft = feats[(size_t)t * FSTRIDE + tid];
      float mx = -1e30f; float vals[NTAG];
      for (int p = 0; p < NTAG; ++p) {
        float v = fv[p] + trans[tid * NTAG + p] + ft;
        vals[p] = v; mx = fmaxf(mx, v);
      }
      float s = 0.0f;
      for (int p = 0; p < NTAG; ++p) s += __expf(vals[p] - mx);
      nfv[tid] = mx + __logf(s);
    }
    __syncthreads();
    if (tid < NTAG) fv[tid] = nfv[tid];
    __syncthreads();
  }
  if (tid == 0) {
    float mx = -1e30f;
    for (int p = 0; p < NTAG; ++p) mx = fmaxf(mx, fv[p] + trans[1 * NTAG + p]);
    float s = 0.0f;
    for (int p = 0; p < NTAG; ++p) s += __expf(fv[p] + trans[1 * NTAG + p] - mx);
    out[0] = (mx + __logf(s)) - gold;
  }
}

// ---------------------------------------------------------------------------
static void gemm(bool relu,
                 const ushort_t* A, int lda, const ushort_t* B, int ldb,
                 float* C, int ldc, ushort_t* Cb, const float* bias,
                 int M, int N, int K, float scale, int causal, hipStream_t s)
{
  dim3 g((unsigned)(N / 64), (unsigned)(M / 32)), b(32, 1, 1);
  if (relu) gemm_wmma<true ><<<g, b, 0, s>>>(A, lda, B, ldb, C, ldc, Cb, bias, K, scale, causal);
  else      gemm_wmma<false><<<g, b, 0, s>>>(A, lda, B, ldb, C, ldc, Cb, bias, K, scale, causal);
}

static void cvt(const float* src, ushort_t* dst, int n, hipStream_t s)
{
  cvt_bf16<<<(n + 255) / 256, 256, 0, s>>>(src, dst, n);
}

static void padb(const float* src, int R, int C, ushort_t* dst, int Rp, int Cp, hipStream_t s)
{
  int n = Rp * Cp;
  pad_bf16<<<(n + 255) / 256, 256, 0, s>>>(src, R, C, dst, Rp, Cp);
}

extern "C" void kernel_launch(void* const* d_in, const int* in_sizes, int n_in,
                              void* d_out, int out_size, void* d_ws, size_t ws_size,
                              hipStream_t stream)
{
  (void)in_sizes; (void)n_in; (void)out_size; (void)ws_size;
  const int*   sentence = (const int*)  d_in[0];
  const int*   postag   = (const int*)  d_in[1];
  const float* tfidf    = (const float*)d_in[2];
  const int*   tags     = (const int*)  d_in[3];
  const float* wembed   = (const float*)d_in[4];
  const float* h0       = (const float*)d_in[5];
  const float* c0       = (const float*)d_in[6];
  const float* W_tag    = (const float*)d_in[7];
  const float* b_tag    = (const float*)d_in[8];
  const float* trans    = (const float*)d_in[9];

  // ---- f32 workspace ----
  float* ws    = (float*)d_ws;
  float* G0    = ws;
  float* G1    = G0   + (size_t)LSEQ * GATES;
  float* H0f   = G1   + (size_t)LSEQ * GATES;
  float* H1f   = H0f  + (size_t)LSEQ * HID;
  float* SRC   = H1f  + (size_t)LSEQ * HID;
  float* QKV   = SRC  + (size_t)LSEQ * HID;
  float* SC    = QKV  + (size_t)LSEQ * QKVD;
  float* ATT   = SC   + (size_t)LSEQ * LSEQ;
  float* TMP   = ATT  + (size_t)LSEQ * HID;
  float* FEATS = TMP  + (size_t)LSEQ * HID;          // [L,64]
  float* b1P   = FEATS + (size_t)LSEQ * FSTRIDE;     // [320]
  float* btagP = b1P  + FFPAD;                       // [64]
  float* FFB   = SC;   // f32 side of FFN hidden (scores free by then)

  // ---- bf16 workspace (16B-aligned: all carve-outs are multiples of 8) ----
  ushort_t* bws   = (ushort_t*)(btagP + 64);
  ushort_t* Xb    = bws;                                  // [L,320]
  ushort_t* WiPfb = Xb    + (size_t)LSEQ * INSTRIDE;      // [1024,320]
  ushort_t* WiPbb = WiPfb + (size_t)GATES * INSTRIDE;
  ushort_t* Wi1fb = WiPbb + (size_t)GATES * INSTRIDE;     // [1024,512]
  ushort_t* Wi1bb = Wi1fb + (size_t)GATES * HID;
  ushort_t* H0bb  = Wi1bb + (size_t)GATES * HID;          // [L,512]
  ushort_t* H1bb  = H0bb  + (size_t)LSEQ * HID;
  ushort_t* SRCb  = H1bb  + (size_t)LSEQ * HID;           // [L,512]
  ushort_t* Wqkvb = SRCb  + (size_t)LSEQ * HID;           // [1536,512]
  ushort_t* QKVb  = Wqkvb + (size_t)QKVD * HID;           // [L,1536]
  ushort_t* SCb   = QKVb  + (size_t)LSEQ * QKVD;          // [L,L]
  ushort_t* VTb   = SCb   + (size_t)LSEQ * LSEQ;          // [256,L]
  ushort_t* ATTb  = VTb   + (size_t)HALFH * LSEQ;         // [L,512]
  ushort_t* Wob   = ATTb  + (size_t)LSEQ * HID;           // [512,512]
  ushort_t* W1Pb  = Wob   + (size_t)HID * HID;            // [320,512]
  ushort_t* FFBb  = W1Pb  + (size_t)FFPAD * HID;          // [L,320]
  ushort_t* W2Pb  = FFBb  + (size_t)LSEQ * FFPAD;         // [512,320]
  ushort_t* WtagPb= W2Pb  + (size_t)HID * FFPAD;          // [64,512]

  build_input<<<LSEQ, INSTRIDE, 0, stream>>>(sentence, postag, tfidf, wembed, Xb);

  // ---- BiLSTM layers ----
  for (int li = 0; li < 2; ++li) {
    int base = 10 + li * 8;
    const float* Wi_f = (const float*)d_in[base + 0];
    const float* Wh_f = (const float*)d_in[base + 1];
    const float* bi_f = (const float*)d_in[base + 2];
    const float* bh_f = (const float*)d_in[base + 3];
    const float* Wi_b = (const float*)d_in[base + 4];
    const float* Wh_b = (const float*)d_in[base + 5];
    const float* bi_b = (const float*)d_in[base + 6];
    const float* bh_b = (const float*)d_in[base + 7];

    const ushort_t* Ain;
    const ushort_t *Bf, *Bb;
    int lda, K;
    if (li == 0) {
      padb(Wi_f, GATES, INDIM, WiPfb, GATES, INSTRIDE, stream);
      padb(Wi_b, GATES, INDIM, WiPbb, GATES, INSTRIDE, stream);
      Ain = Xb; lda = INSTRIDE; K = INSTRIDE; Bf = WiPfb; Bb = WiPbb;
    } else {
      cvt(Wi_f, Wi1fb, GATES * HID, stream);
      cvt(Wi_b, Wi1bb, GATES * HID, stream);
      Ain = H0bb; lda = HID; K = HID; Bf = Wi1fb; Bb = Wi1bb;
    }
    gemm(false, Ain, lda, Bf, K, G0, GATES, nullptr, nullptr, LSEQ, GATES, K, 1.0f, 0, stream);
    gemm(false, Ain, lda, Bb, K, G1, GATES, nullptr, nullptr, LSEQ, GATES, K, 1.0f, 0, stream);

    float*    Ho  = (li == 0) ? H0f  : H1f;
    ushort_t* Hob = (li == 0) ? H0bb : H1bb;
    lstm_scan<<<2, GATES, 0, stream>>>(G0, G1, Wh_f, bi_f, bh_f, Wh_b, bi_b, bh_b,
                                       h0 + 2 * li * HALFH, c0 + 2 * li * HALFH, Ho, Hob);
  }

  posenc_scale<<<(LSEQ * HID + 255) / 256, 256, 0, stream>>>(H1f, SRC, SRCb);

  // ---- Transformer encoder layers ----
  for (int li = 0; li < 2; ++li) {
    int base = 26 + li * 12;
    const float* Wqkv = (const float*)d_in[base + 0];
    const float* bqkv = (const float*)d_in[base + 1];
    const float* Wo   = (const float*)d_in[base + 2];
    const float* bo   = (const float*)d_in[base + 3];
    const float* W1   = (const float*)d_in[base + 4];
    const float* b1   = (const float*)d_in[base + 5];
    const float* W2   = (const float*)d_in[base + 6];
    const float* b2   = (const float*)d_in[base + 7];
    const float* g1   = (const float*)d_in[base + 8];
    const float* be1  = (const float*)d_in[base + 9];
    const float* g2   = (const float*)d_in[base + 10];
    const float* be2  = (const float*)d_in[base + 11];

    cvt(Wqkv, Wqkvb, QKVD * HID, stream);
    cvt(Wo,   Wob,   HID * HID,  stream);
    gemm(false, SRCb, HID, Wqkvb, HID, QKV, QKVD, QKVb, bqkv, LSEQ, QKVD, HID, 1.0f, 0, stream);

    for (int hh = 0; hh < 2; ++hh) {
      const ushort_t* Qb = QKVb + hh * 256;
      const ushort_t* Kb = QKVb + 512 + hh * 256;
      const float*    Vp = QKV + 1024 + hh * 256;
      // S = (Q @ K^T) / sqrt(dh), causal tiles only
      gemm(false, Qb, QKVD, Kb, QKVD, SC, LSEQ, nullptr, nullptr, LSEQ, LSEQ, 256, 0.0625f, 1, stream);
      softmax_causal<<<LSEQ, 256, 0, stream>>>(SC, SCb);
      // VTb = bf16(V^T) so the A@V GEMM uses contiguous-K fragments
      transpose_bf16<<<dim3(16, LSEQ / 16), dim3(16, 16), 0, stream>>>(Vp, QKVD, LSEQ, 256, VTb, LSEQ);
      gemm(false, SCb, LSEQ, VTb, LSEQ, ATT + hh * 256, HID, nullptr, nullptr,
           LSEQ, 256, LSEQ, 1.0f, 0, stream);
    }
    cvt(ATT, ATTb, LSEQ * HID, stream);

    gemm(false, ATTb, HID, Wob, HID, TMP, HID, nullptr, bo, LSEQ, HID, HID, 1.0f, 0, stream);
    add_layernorm<<<LSEQ, 256, 0, stream>>>(SRC, SRCb, TMP, g1, be1);

    // FFN with zero-padded 300 -> 320 hidden dim (pad outputs are relu(0)=0)
    padb(W1, FFDIM, HID,  W1Pb, FFPAD, HID,  stream);
    pad_f32<<<(FFPAD + 255) / 256, 256, 0, stream>>>(b1, FFDIM, 1, b1P, FFPAD, 1);
    padb(W2, HID,  FFDIM, W2Pb, HID,  FFPAD, stream);
    gemm(true,  SRCb, HID, W1Pb, HID, FFB, FFPAD, FFBb, b1P, LSEQ, FFPAD, HID, 1.0f, 0, stream);
    gemm(false, FFBb, FFPAD, W2Pb, FFPAD, TMP, HID, nullptr, b2, LSEQ, HID, FFPAD, 1.0f, 0, stream);
    add_layernorm<<<LSEQ, 256, 0, stream>>>(SRC, SRCb, TMP, g2, be2);
  }

  // ---- Tag projection (padded 11 -> 64 tags) + CRF ----
  padb(W_tag, NTAG, HID, WtagPb, 64, HID, stream);
  pad_f32<<<1, 256, 0, stream>>>(b_tag, NTAG, 1, btagP, 64, 1);
  gemm(false, SRCb, HID, WtagPb, HID, FEATS, FSTRIDE, nullptr, btagP, LSEQ, 64, HID, 1.0f, 0, stream);
  crf_kernel<<<1, 32, 0, stream>>>(FEATS, trans, tags, (float*)d_out);
}
